// Net_49804440764839
// MI455X (gfx1250) — matrix operand order repaired and checked by reading, not compiled, not run on hardware
//
#include <hip/hip_runtime.h>
#include <cmath>

// ---------------------------------------------------------------------------
// MI455X (gfx1250) implementation: bf16 WMMA GEMMs + flash attention,
// persistent-block GRUs, small FC head.  wave32 everywhere.
// ---------------------------------------------------------------------------

typedef __attribute__((ext_vector_type(16))) __bf16 v16bf;
typedef __attribute__((ext_vector_type(8)))  __bf16 v8bf;
typedef __attribute__((ext_vector_type(8)))  float  v8f;

#define SEQLEN 4096
#define DREAL  300
#define DPAD   320
#define H3     384
#define HID    128

// ---- WMMA fragment helpers (layouts per CDNA5 ISA 7.12.2) ------------------
__device__ __forceinline__ v8f wmma_bf16(v16bf a, v16bf b, v8f c) {
  return __builtin_amdgcn_wmma_f32_16x16x32_bf16(false, a, false, b,
                                                 (short)0, c, false, false);
}

// A-matrix 16x32 bf16: lanes0-15 hold K{0..7,16..23}, lanes16-31 K{8..15,24..31}
__device__ __forceinline__ v16bf load_a_frag(const __bf16* rowbase, int lane) {
  int hi = (lane >> 4) & 1;
  const __bf16* p = rowbase + hi * 8;
  v8bf lo = *(const v8bf*)p;
  v8bf hh = *(const v8bf*)(p + 16);
  v16bf r;
#pragma unroll
  for (int i = 0; i < 8; ++i) { r[i] = lo[i]; r[i + 8] = hh[i]; }
  return r;
}

// B-matrix 32x16 bf16 (as rows of B^T): lanes0-15 K{0..7,16..23}, lanes16-31 K{8..15,24..31}
__device__ __forceinline__ v16bf load_b_frag(const __bf16* rowbase, int lane) {
  int hi = (lane >> 4) & 1;
  const __bf16* p = rowbase + hi * 8;
  v8bf lo = *(const v8bf*)p;
  v8bf hh = *(const v8bf*)(p + 16);
  v16bf r;
#pragma unroll
  for (int i = 0; i < 8; ++i) { r[i] = lo[i]; r[i + 8] = hh[i]; }
  return r;
}

// ---- pack fp32 -> zero-padded bf16 ----------------------------------------
__global__ __launch_bounds__(256) void pack_bf16_kernel(
    const float* __restrict__ src, __bf16* __restrict__ dst,
    int R, int C, int Rp, int Cp) {
  int idx = blockIdx.x * blockDim.x + threadIdx.x;
  if (idx >= Rp * Cp) return;
  int r = idx / Cp, c = idx - r * Cp;
  dst[idx] = (r < R && c < C) ? (__bf16)src[r * C + c] : (__bf16)0.0f;
}

// ---- GEMM: C[M,N] = A[M,Kp] @ B[N,Kp]^T + bias; out fp32 or bf16 ----------
// One 32x32 tile per wave, K in steps of 32 (all dims padded to /32).
__global__ __launch_bounds__(256) void gemm_bf16_kernel(
    const __bf16* __restrict__ A, int lda,
    const __bf16* __restrict__ B, int ldb,
    const float* __restrict__ bias, int realN,
    float* __restrict__ outF, __bf16* __restrict__ outB, int ldc,
    int M, int N, int K) {
  int lane = threadIdx.x & 31;
  int wave = threadIdx.x >> 5;
  int tilesN = N >> 5;
  int tile = blockIdx.x * (blockDim.x >> 5) + wave;
  if (tile >= (M >> 5) * tilesN) return;           // wave-uniform
  int row0 = (tile / tilesN) << 5;
  int col0 = (tile % tilesN) << 5;

  v8f acc00 = {}, acc01 = {}, acc10 = {}, acc11 = {};
  int ar = row0 + (lane & 15);
  int br = col0 + (lane & 15);
  for (int k0 = 0; k0 < K; k0 += 32) {
    v16bf a0 = load_a_frag(A + (size_t)ar * lda + k0, lane);
    v16bf a1 = load_a_frag(A + (size_t)(ar + 16) * lda + k0, lane);
    v16bf b0 = load_b_frag(B + (size_t)br * ldb + k0, lane);
    v16bf b1 = load_b_frag(B + (size_t)(br + 16) * ldb + k0, lane);
    acc00 = wmma_bf16(a0, b0, acc00);
    acc01 = wmma_bf16(a0, b1, acc01);
    acc10 = wmma_bf16(a1, b0, acc10);
    acc11 = wmma_bf16(a1, b1, acc11);
  }
  int cn = lane & 15;
  int rb = (lane >> 4) << 3;
  v8f accs[2][2] = {{acc00, acc01}, {acc10, acc11}};
#pragma unroll
  for (int i = 0; i < 2; ++i) {
#pragma unroll
    for (int j = 0; j < 2; ++j) {
      int col = col0 + j * 16 + cn;
      float bv = (col < realN) ? bias[col] : 0.0f;
#pragma unroll
      for (int r = 0; r < 8; ++r) {
        int row = row0 + i * 16 + rb + r;
        float v = accs[i][j][r] + bv;
        if (outB) outB[(size_t)row * ldc + col] = (__bf16)v;
        else      outF[(size_t)row * ldc + col] = v;
      }
    }
  }
}

// ---- Flash attention: O = softmax(Q K^T * scale) V -------------------------
// 128 threads (4 waves), 64 query rows per block, key blocks of 64.
// LDS: Qs[64][320] | Ks[64][320] | Vt[320][72 padded] | Ps[4][16][64]
#define FA_QS   0
#define FA_KS   40960
#define FA_VT   81920
#define FA_PS   (81920 + 46080)
#define FA_SMEM (FA_PS + 8192)

__global__ __launch_bounds__(128) void flash_attn_kernel(
    const __bf16* __restrict__ Q, const __bf16* __restrict__ Kmat,
    const __bf16* __restrict__ V, float* __restrict__ O, float scale) {
  extern __shared__ char smem[];
  __bf16* Qs = (__bf16*)(smem + FA_QS);
  __bf16* Ks = (__bf16*)(smem + FA_KS);
  __bf16* Vt = (__bf16*)(smem + FA_VT);   // [320][72]
  __bf16* Ps = (__bf16*)(smem + FA_PS);   // per-wave [16][64]

  const int tid  = threadIdx.x;
  const int lane = tid & 31;
  const int wave = tid >> 5;
  const int r0   = blockIdx.x * 64;

  { // load Q block
    const uint4* Qg = (const uint4*)(Q + (size_t)r0 * DPAD);
    uint4* Qs4 = (uint4*)Qs;
    for (int i = tid; i < 64 * 40; i += 128) Qs4[i] = Qg[i];
  }

  v8f o[20];
#pragma unroll
  for (int t = 0; t < 20; ++t) o[t] = (v8f){};
  float m_r[8], l_r[8];
#pragma unroll
  for (int r = 0; r < 8; ++r) { m_r[r] = -1e30f; l_r[r] = 0.0f; }

  const int lrow = wave * 16 + (lane & 15);
  __bf16* Pw = Ps + wave * (16 * 64);

  for (int kb = 0; kb < SEQLEN / 64; ++kb) {
    __syncthreads();  // protect LDS reuse (also covers initial Q fill)
    { // stage K block and transposed V block
      const uint4* Kg = (const uint4*)(Kmat + (size_t)kb * 64 * DPAD);
      uint4* Ks4 = (uint4*)Ks;
      for (int i = tid; i < 64 * 40; i += 128) Ks4[i] = Kg[i];
      for (int i = tid; i < 64 * 40; i += 128) {
        int kk = i / 40;
        int c8 = (i - kk * 40) * 8;
        v8bf vv = *(const v8bf*)(V + ((size_t)(kb * 64 + kk)) * DPAD + c8);
#pragma unroll
        for (int e = 0; e < 8; ++e) Vt[(c8 + e) * 72 + kk] = vv[e];
      }
    }
    __syncthreads();

    // S tile: 16 rows x 64 cols per wave
    v8f s[4];
#pragma unroll
    for (int j = 0; j < 4; ++j) s[j] = (v8f){};
    for (int k0 = 0; k0 < DPAD; k0 += 32) {
      v16bf aq = load_a_frag(Qs + lrow * DPAD + k0, lane);
#pragma unroll
      for (int j = 0; j < 4; ++j) {
        v16bf bk = load_b_frag(Ks + (j * 16 + (lane & 15)) * DPAD + k0, lane);
        s[j] = wmma_bf16(aq, bk, s[j]);
      }
    }
#pragma unroll
    for (int j = 0; j < 4; ++j)
#pragma unroll
      for (int r = 0; r < 8; ++r) s[j][r] *= scale;

    // online softmax (rows replicated across 16-lane halves)
#pragma unroll
    for (int r = 0; r < 8; ++r) {
      float mx = fmaxf(fmaxf(s[0][r], s[1][r]), fmaxf(s[2][r], s[3][r]));
      mx = fmaxf(mx, __shfl_xor(mx, 1));
      mx = fmaxf(mx, __shfl_xor(mx, 2));
      mx = fmaxf(mx, __shfl_xor(mx, 4));
      mx = fmaxf(mx, __shfl_xor(mx, 8));
      float mnew  = fmaxf(m_r[r], mx);
      float alpha = __expf(m_r[r] - mnew);
      m_r[r] = mnew;
      float rs = 0.0f;
#pragma unroll
      for (int j = 0; j < 4; ++j) {
        float pv = __expf(s[j][r] - mnew);
        s[j][r] = pv;
        rs += pv;
      }
      rs += __shfl_xor(rs, 1);
      rs += __shfl_xor(rs, 2);
      rs += __shfl_xor(rs, 4);
      rs += __shfl_xor(rs, 8);
      l_r[r] = l_r[r] * alpha + rs;
#pragma unroll
      for (int t = 0; t < 20; ++t) o[t][r] *= alpha;
    }

    // stage P (C-layout -> A-layout via per-wave LDS tile)
    {
      int rb = (lane >> 4) << 3;
      int cn = lane & 15;
#pragma unroll
      for (int j = 0; j < 4; ++j)
#pragma unroll
        for (int r = 0; r < 8; ++r)
          Pw[(rb + r) * 64 + j * 16 + cn] = (__bf16)s[j][r];
    }

    // O += P @ V  (K = 64, N = 320)
#pragma unroll
    for (int kc = 0; kc < 64; kc += 32) {
      v16bf ap = load_a_frag(Pw + (lane & 15) * 64 + kc, lane);
#pragma unroll
      for (int t = 0; t < 20; ++t) {
        v16bf bv = load_b_frag(Vt + (t * 16 + (lane & 15)) * 72 + kc, lane);
        o[t] = wmma_bf16(ap, bv, o[t]);
      }
    }
  }

  // normalize + store
  float inv[8];
#pragma unroll
  for (int r = 0; r < 8; ++r) inv[r] = 1.0f / l_r[r];
  int rowb = r0 + wave * 16 + ((lane >> 4) << 3);
  int cn = lane & 15;
#pragma unroll
  for (int t = 0; t < 20; ++t)
#pragma unroll
    for (int r = 0; r < 8; ++r)
      O[(size_t)(rowb + r) * DPAD + t * 16 + cn] = o[t][r] * inv[r];
}

// ---- LayerNorm(X + residual) -> bf16 padded; one wave per row --------------
__global__ __launch_bounds__(256) void ln_residual_kernel(
    const float* __restrict__ X, int ldx,
    const void* __restrict__ R, int ldr, int r_bf16,
    const float* __restrict__ g, const float* __restrict__ b,
    __bf16* __restrict__ out, int ldo) {
  int lane = threadIdx.x & 31;
  int row = blockIdx.x * (blockDim.x >> 5) + (threadIdx.x >> 5);
  if (row >= SEQLEN) return;
  const float* xr = X + (size_t)row * ldx;
  float vals[10];
  float s = 0.0f, ss = 0.0f;
#pragma unroll
  for (int i = 0; i < 10; ++i) {
    int c = lane + i * 32;
    float v = 0.0f;
    if (c < DREAL) {
      float rv = r_bf16 ? (float)((const __bf16*)R)[(size_t)row * ldr + c]
                        : ((const float*)R)[(size_t)row * ldr + c];
      v = xr[c] + rv;
      s += v; ss += v * v;
    }
    vals[i] = v;
  }
#pragma unroll
  for (int m = 1; m <= 16; m <<= 1) { s += __shfl_xor(s, m); ss += __shfl_xor(ss, m); }
  float mean = s * (1.0f / DREAL);
  float var  = ss * (1.0f / DREAL) - mean * mean;
  float rinv = rsqrtf(var + 1e-5f);
#pragma unroll
  for (int i = 0; i < 10; ++i) {
    int c = lane + i * 32;
    if (c < DREAL)
      out[(size_t)row * ldo + c] = (__bf16)(g[c] * (vals[i] - mean) * rinv + b[c]);
    else if (c < DPAD)
      out[(size_t)row * ldo + c] = (__bf16)0.0f;
  }
}

// ---- GRU scan: one block per sequence, W_hh register-resident --------------
__device__ __forceinline__ float sigmoidf_(float x) { return 1.0f / (1.0f + __expf(-x)); }

__global__ __launch_bounds__(384) void gru_kernel(
    const float* __restrict__ gx1, const float* __restrict__ gx2,
    const float* __restrict__ whh1, const float* __restrict__ whh2,
    const float* __restrict__ bhh1, const float* __restrict__ bhh2,
    float* __restrict__ hout) {
  const float* gx  = blockIdx.x ? gx2  : gx1;
  const float* whh = blockIdx.x ? whh2 : whh1;
  const float* bhh = blockIdx.x ? bhh2 : bhh1;
  float* ho = hout + blockIdx.x * HID;

  __shared__ float hs[HID];
  __shared__ float ghs[H3];
  int i = threadIdx.x;  // 384
  float w[HID];
#pragma unroll
  for (int j = 0; j < HID; ++j) w[j] = whh[(size_t)i * HID + j];
  float bi = bhh[i];
  if (i < HID) hs[i] = 0.0f;
  __syncthreads();

  for (int t = 0; t < SEQLEN; ++t) {
    float acc = bi;
#pragma unroll
    for (int j = 0; j < HID; ++j) acc += w[j] * hs[j];
    ghs[i] = acc;
    __syncthreads();
    if (i < HID) {
      const float* g = gx + (size_t)t * H3;
      float r = sigmoidf_(g[i] + ghs[i]);
      float z = sigmoidf_(g[i + HID] + ghs[i + HID]);
      float n = tanhf(g[i + 2 * HID] + r * ghs[i + 2 * HID]);
      hs[i] = (1.0f - z) * n + z * hs[i];
    }
    __syncthreads();
  }
  if (i < HID) ho[i] = hs[i];
}

// ---- FC head: relu(fc1) -> fc2 -> log_softmax ------------------------------
__global__ __launch_bounds__(256) void fc_head_kernel(
    const float* __restrict__ hvec,
    const float* __restrict__ fc1w, const float* __restrict__ fc1b,
    const float* __restrict__ fc2w, const float* __restrict__ fc2b,
    float* __restrict__ out) {
  __shared__ float hc[256];
  __shared__ float y[256];
  int i = threadIdx.x;
  hc[i] = hvec[i];
  __syncthreads();
  float acc = fc1b[i];
  for (int j = 0; j < 256; ++j) acc += fc1w[(size_t)i * 256 + j] * hc[j];
  y[i] = fmaxf(acc, 0.0f);
  __syncthreads();
  if (i == 0) {
    float z[3];
    for (int o2 = 0; o2 < 3; ++o2) {
      float a = fc2b[o2];
      for (int j = 0; j < 256; ++j) a += fc2w[(size_t)o2 * 256 + j] * y[j];
      z[o2] = a;
    }
    float mx = fmaxf(z[0], fmaxf(z[1], z[2]));
    float se = __expf(z[0] - mx) + __expf(z[1] - mx) + __expf(z[2] - mx);
    float lse = mx + logf(se);
    out[0] = z[0] - lse; out[1] = z[1] - lse; out[2] = z[2] - lse;
  }
}

// ---------------------------------------------------------------------------
extern "C" void kernel_launch(void* const* d_in, const int* in_sizes, int n_in,
                              void* d_out, int out_size, void* d_ws, size_t ws_size,
                              hipStream_t stream) {
  const float* x1     = (const float*)d_in[0];
  const float* x2     = (const float*)d_in[1];
  const float* q_w    = (const float*)d_in[2];
  const float* q_b    = (const float*)d_in[3];
  const float* k_w    = (const float*)d_in[4];
  const float* k_b    = (const float*)d_in[5];
  const float* v_w    = (const float*)d_in[6];
  const float* v_b    = (const float*)d_in[7];
  const float* aln_g  = (const float*)d_in[8];
  const float* aln_b  = (const float*)d_in[9];
  const float* w_w    = (const float*)d_in[10];
  const float* w_b    = (const float*)d_in[11];
  const float* mln_g  = (const float*)d_in[12];
  const float* mln_b  = (const float*)d_in[13];
  const float* g1_wih = (const float*)d_in[14];
  const float* g1_whh = (const float*)d_in[15];
  const float* g1_bih = (const float*)d_in[16];
  const float* g1_bhh = (const float*)d_in[17];
  const float* g2_wih = (const float*)d_in[18];
  const float* g2_whh = (const float*)d_in[19];
  const float* g2_bih = (const float*)d_in[20];
  const float* g2_bhh = (const float*)d_in[21];
  const float* fc1_w  = (const float*)d_in[22];
  const float* fc1_b  = (const float*)d_in[23];
  const float* fc2_w  = (const float*)d_in[24];
  const float* fc2_b  = (const float*)d_in[25];

  char* base = (char*)d_ws;
  size_t off = 0;
  auto carve = [&](size_t bytes) -> void* {
    void* p = base + off;
    off += (bytes + 255) & ~(size_t)255;
    return p;
  };

  const size_t szBF = (size_t)SEQLEN * DPAD * sizeof(__bf16);
  __bf16* outBuf[2] = {(__bf16*)carve(szBF), (__bf16*)carve(szBF)};
  __bf16* Qb = (__bf16*)carve(szBF);
  __bf16* Kb = (__bf16*)carve(szBF);
  __bf16* Vb = (__bf16*)carve(szBF);
  __bf16* Hb = (__bf16*)carve(szBF);
  __bf16* Wq = (__bf16*)carve((size_t)DPAD * DPAD * 2);
  __bf16* Wk = (__bf16*)carve((size_t)DPAD * DPAD * 2);
  __bf16* Wv = (__bf16*)carve((size_t)DPAD * DPAD * 2);
  __bf16* Ww = (__bf16*)carve((size_t)DPAD * DPAD * 2);
  __bf16* Wih[2] = {(__bf16*)carve((size_t)H3 * DPAD * 2),
                    (__bf16*)carve((size_t)H3 * DPAD * 2)};
  float* Oacc = (float*)carve((size_t)SEQLEN * DPAD * 4);
  float* O2   = (float*)carve((size_t)SEQLEN * DPAD * 4);
  float* gx[2] = {(float*)carve((size_t)SEQLEN * H3 * 4),
                  (float*)carve((size_t)SEQLEN * H3 * 4)};
  float* hcat = (float*)carve(256 * 4);

  const float attn_scale = 1.0f / sqrtf((float)DREAL);
  const float* xin[2] = {x1, x2};

  // ---- pack inputs / weights to padded bf16
  {
    int n = SEQLEN * DPAD;
    pack_bf16_kernel<<<(n + 255) / 256, 256, 0, stream>>>(x1, outBuf[0], SEQLEN, DREAL, SEQLEN, DPAD);
    pack_bf16_kernel<<<(n + 255) / 256, 256, 0, stream>>>(x2, outBuf[1], SEQLEN, DREAL, SEQLEN, DPAD);
    int nw = DPAD * DPAD;
    pack_bf16_kernel<<<(nw + 255) / 256, 256, 0, stream>>>(q_w, Wq, DREAL, DREAL, DPAD, DPAD);
    pack_bf16_kernel<<<(nw + 255) / 256, 256, 0, stream>>>(k_w, Wk, DREAL, DREAL, DPAD, DPAD);
    pack_bf16_kernel<<<(nw + 255) / 256, 256, 0, stream>>>(v_w, Wv, DREAL, DREAL, DPAD, DPAD);
    pack_bf16_kernel<<<(nw + 255) / 256, 256, 0, stream>>>(w_w, Ww, DREAL, DREAL, DPAD, DPAD);
    int ni = H3 * DPAD;
    pack_bf16_kernel<<<(ni + 255) / 256, 256, 0, stream>>>(g1_wih, Wih[0], H3, DREAL, H3, DPAD);
    pack_bf16_kernel<<<(ni + 255) / 256, 256, 0, stream>>>(g2_wih, Wih[1], H3, DREAL, H3, DPAD);
  }

  const int gemmBlocks320 = (SEQLEN / 32) * (DPAD / 32) / 8;  // 160
  const int gemmBlocks384 = (SEQLEN / 32) * (H3 / 32) / 8;    // 192
  const int lnBlocks = SEQLEN / 8;                            // 512

  for (int s = 0; s < 2; ++s) {
    for (int st = 0; st < 3; ++st) {
      gemm_bf16_kernel<<<gemmBlocks320, 256, 0, stream>>>(
          outBuf[s], DPAD, Wq, DPAD, q_b, DREAL, nullptr, Qb, DPAD, SEQLEN, DPAD, DPAD);
      gemm_bf16_kernel<<<gemmBlocks320, 256, 0, stream>>>(
          outBuf[s], DPAD, Wk, DPAD, k_b, DREAL, nullptr, Kb, DPAD, SEQLEN, DPAD, DPAD);
      gemm_bf16_kernel<<<gemmBlocks320, 256, 0, stream>>>(
          outBuf[s], DPAD, Wv, DPAD, v_b, DREAL, nullptr, Vb, DPAD, SEQLEN, DPAD, DPAD);

      flash_attn_kernel<<<SEQLEN / 64, 128, FA_SMEM, stream>>>(Qb, Kb, Vb, Oacc, attn_scale);

      ln_residual_kernel<<<lnBlocks, 256, 0, stream>>>(
          Oacc, DPAD, outBuf[s], DPAD, 1, aln_g, aln_b, Hb, DPAD);

      gemm_bf16_kernel<<<gemmBlocks320, 256, 0, stream>>>(
          Hb, DPAD, Ww, DPAD, w_b, DREAL, O2, nullptr, DPAD, SEQLEN, DPAD, DPAD);

      ln_residual_kernel<<<lnBlocks, 256, 0, stream>>>(
          O2, DPAD, xin[s], DREAL, 0, mln_g, mln_b, outBuf[s], DPAD);
    }
  }

  // GRU input projections: gx = a @ W_ih^T + b_ih
  gemm_bf16_kernel<<<gemmBlocks384, 256, 0, stream>>>(
      outBuf[0], DPAD, Wih[0], DPAD, g1_bih, H3, gx[0], nullptr, H3, SEQLEN, H3, DPAD);
  gemm_bf16_kernel<<<gemmBlocks384, 256, 0, stream>>>(
      outBuf[1], DPAD, Wih[1], DPAD, g2_bih, H3, gx[1], nullptr, H3, SEQLEN, H3, DPAD);

  gru_kernel<<<2, H3, 0, stream>>>(gx[0], gx[1], g1_whh, g2_whh, g1_bhh, g2_bhh, hcat);

  fc_head_kernel<<<1, 256, 0, stream>>>(hcat, fc1_w, fc1_b, fc2_w, fc2_b, (float*)d_out);
}
